// InvertibleDBNN_36163624632835
// MI455X (gfx1250) — matrix operand order repaired and checked by reading, not compiled, not run on hardware
//
#include <hip/hip_runtime.h>
#include <stdint.h>

// Problem constants (match reference)
#define BATCH 65536
#define NCLS  128          // C = GEMM K
#define FDIM  256          // F = GEMM N
#define MT    64           // batch rows per chunk per block
#define WST   136          // padded K-stride (bf16 elems) in LDS: 128 + 8 -> conflict-free b128 reads
#define LNEPS 1e-5f

typedef __attribute__((ext_vector_type(16))) __bf16 v16bf;
typedef __attribute__((ext_vector_type(8)))  float  v8f;

typedef union { v16bf v; uint32_t u[8]; uint4 q[2]; } BfOp;

__device__ __forceinline__ uint16_t f2bf(float x) {   // round-to-nearest-even fp32 -> bf16
  uint32_t u = __float_as_uint(x);
  u += 0x7FFFu + ((u >> 16) & 1u);
  return (uint16_t)(u >> 16);
}
__device__ __forceinline__ float bf2f(uint16_t h) {
  return __uint_as_float(((uint32_t)h) << 16);
}

// ---------------------------------------------------------------------------
// Prep 1: extract the (C,F) diagonals of the (C*F, F) weights, split fp32 into
// bf16 hi/lo, store transposed as (F, C) so the WMMA B operand is K-contiguous.
// ---------------------------------------------------------------------------
__global__ void prep_diag(const float* __restrict__ wl, const float* __restrict__ wn,
                          uint16_t* __restrict__ lhi, uint16_t* __restrict__ llo,
                          uint16_t* __restrict__ nhi, uint16_t* __restrict__ nlo) {
  const int c = blockIdx.x;      // 0..127
  const int f = threadIdx.x;     // 0..255
  const size_t idx = (size_t)c * (FDIM * FDIM) + (size_t)f * (FDIM + 1);
  const float a = wl[idx];
  const float b = wn[idx];
  const uint16_t ah = f2bf(a); const uint16_t al = f2bf(a - bf2f(ah));
  const uint16_t bh = f2bf(b); const uint16_t bl = f2bf(b - bf2f(bh));
  const int o = f * NCLS + c;    // transposed (F x C)
  lhi[o] = ah; llo[o] = al; nhi[o] = bh; nlo[o] = bl;
}

// ---------------------------------------------------------------------------
// Prep 2: softmax(feature_attention) and combined bias.
// ---------------------------------------------------------------------------
__global__ void prep_vec(const float* __restrict__ fa, const float* __restrict__ bl,
                         const float* __restrict__ bn,
                         float* __restrict__ attn, float* __restrict__ biasc) {
  __shared__ float sb[FDIM];
  const int t = threadIdx.x;
  const float v = fa[t];
  sb[t] = v; __syncthreads();
  for (int s = FDIM / 2; s > 0; s >>= 1) {
    if (t < s) sb[t] = fmaxf(sb[t], sb[t + s]);
    __syncthreads();
  }
  const float mx = sb[0]; __syncthreads();
  const float e = __expf(v - mx);
  sb[t] = e; __syncthreads();
  for (int s = FDIM / 2; s > 0; s >>= 1) {
    if (t < s) sb[t] += sb[t + s];
    __syncthreads();
  }
  attn[t]  = e / sb[0];
  biasc[t] = bl[t] + bn[t];
}

// ---------------------------------------------------------------------------
// Main: persistent GEMM + epilogue + layernorm.
// Block = 256 threads = 8 wave32. Wave w: m_tile = w>>1 (16 rows), n_half = w&1
// (8 of 16 N-tiles). 3-term bf16 split per weight matrix:
//   acc += Ahi*Whi + Alo*Whi + Ahi*Wlo        (near-fp32 accuracy)
// All split diag-weights live in LDS (padded stride) for the kernel lifetime.
// ---------------------------------------------------------------------------
__global__ __launch_bounds__(256) void dbnn_main(
    const float* __restrict__ probs,
    const uint16_t* __restrict__ gLH, const uint16_t* __restrict__ gLL,
    const uint16_t* __restrict__ gNH, const uint16_t* __restrict__ gNL,
    const float* __restrict__ attn, const float* __restrict__ biasc,
    const float* __restrict__ gamma, const float* __restrict__ beta,
    float* __restrict__ out, int nchunks)
{
  extern __shared__ char smem[];
  uint16_t* wLH = (uint16_t*)smem;            // [FDIM][WST]
  uint16_t* wLL = wLH + FDIM * WST;
  uint16_t* wNH = wLL + FDIM * WST;
  uint16_t* wNL = wNH + FDIM * WST;
  uint16_t* aHi = wNL + FDIM * WST;           // [MT][WST]
  uint16_t* aLo = aHi + MT * WST;
  float*    redS = (float*)(aLo + MT * WST);  // [2][MT]
  float*    redQ = redS + 2 * MT;

  const int t = threadIdx.x;

  // Stage all split diagonal weights in LDS (uint32 = 2 bf16 at a time).
  for (int i = t; i < (FDIM * NCLS) / 2; i += 256) {
    const int col = i >> 6;              // each col has 64 uint32 (128 bf16)
    const int k   = (i & 63) * 2;
    const int o   = col * WST + k;
    *(uint32_t*)&wLH[o] = ((const uint32_t*)gLH)[i];
    *(uint32_t*)&wLL[o] = ((const uint32_t*)gLL)[i];
    *(uint32_t*)&wNH[o] = ((const uint32_t*)gNH)[i];
    *(uint32_t*)&wNL[o] = ((const uint32_t*)gNL)[i];
  }
  __syncthreads();

  const int wave   = t >> 5;
  const int lane   = t & 31;
  const int m_tile = wave >> 1;     // 0..3
  const int n_half = wave & 1;      // 0..1
  const int lh     = lane & 15;
  const int hh     = lane >> 4;
  const int arow   = m_tile * 16 + lh;

  for (int chunk = blockIdx.x; chunk < nchunks; chunk += gridDim.x) {
    const int rowbase = chunk * MT;

    // Load 64x128 fp32 A chunk, split into bf16 hi/lo in LDS.
    for (int i = t; i < (MT * NCLS) / 4; i += 256) {
      const int row = i >> 5;
      const int k4  = (i & 31) * 4;
      const float4 v = *(const float4*)(probs + (size_t)(rowbase + row) * NCLS + k4);
      const uint16_t h0 = f2bf(v.x), h1 = f2bf(v.y), h2 = f2bf(v.z), h3 = f2bf(v.w);
      const uint16_t l0 = f2bf(v.x - bf2f(h0)), l1 = f2bf(v.y - bf2f(h1));
      const uint16_t l2 = f2bf(v.z - bf2f(h2)), l3 = f2bf(v.w - bf2f(h3));
      uint2 ph, pl;
      ph.x = (uint32_t)h0 | ((uint32_t)h1 << 16); ph.y = (uint32_t)h2 | ((uint32_t)h3 << 16);
      pl.x = (uint32_t)l0 | ((uint32_t)l1 << 16); pl.y = (uint32_t)l2 | ((uint32_t)l3 << 16);
      *(uint2*)&aHi[row * WST + k4] = ph;
      *(uint2*)&aLo[row * WST + k4] = pl;
    }
    __syncthreads();

    v8f accL[8], accN[8];
    #pragma unroll
    for (int j = 0; j < 8; ++j)
      #pragma unroll
      for (int e = 0; e < 8; ++e) { accL[j][e] = 0.f; accN[j][e] = 0.f; }

    for (int ks = 0; ks < 4; ++ks) {
      // A operand (16-bit A 16x32 layout): lanes<16 hold K {k0..k0+7, k0+16..k0+23}
      const int k0 = ks * 32 + hh * 8;
      BfOp aH, aL;
      aH.q[0] = *(const uint4*)&aHi[arow * WST + k0];
      aH.q[1] = *(const uint4*)&aHi[arow * WST + k0 + 16];
      aL.q[0] = *(const uint4*)&aLo[arow * WST + k0];
      aL.q[1] = *(const uint4*)&aLo[arow * WST + k0 + 16];
      // B operand (32x16): lane n<16 -> col n, K {kb..kb+7, kb+16..kb+23}
      const int kb = ks * 32 + hh * 8;
      #pragma unroll
      for (int j = 0; j < 8; ++j) {
        const int col = (n_half * 8 + j) * 16 + lh;
        const int o   = col * WST + kb;
        BfOp bLH, bLL, bNH, bNL;
        bLH.q[0] = *(const uint4*)&wLH[o]; bLH.q[1] = *(const uint4*)&wLH[o + 16];
        bLL.q[0] = *(const uint4*)&wLL[o]; bLL.q[1] = *(const uint4*)&wLL[o + 16];
        bNH.q[0] = *(const uint4*)&wNH[o]; bNH.q[1] = *(const uint4*)&wNH[o + 16];
        bNL.q[0] = *(const uint4*)&wNL[o]; bNL.q[1] = *(const uint4*)&wNL[o + 16];
        accL[j] = __builtin_amdgcn_wmma_f32_16x16x32_bf16(false, aH.v, false, bLH.v, (short)0, accL[j], false, false);
        accL[j] = __builtin_amdgcn_wmma_f32_16x16x32_bf16(false, aL.v, false, bLH.v, (short)0, accL[j], false, false);
        accL[j] = __builtin_amdgcn_wmma_f32_16x16x32_bf16(false, aH.v, false, bLL.v, (short)0, accL[j], false, false);
        accN[j] = __builtin_amdgcn_wmma_f32_16x16x32_bf16(false, aH.v, false, bNH.v, (short)0, accN[j], false, false);
        accN[j] = __builtin_amdgcn_wmma_f32_16x16x32_bf16(false, aL.v, false, bNH.v, (short)0, accN[j], false, false);
        accN[j] = __builtin_amdgcn_wmma_f32_16x16x32_bf16(false, aH.v, false, bNL.v, (short)0, accN[j], false, false);
      }
    }

    // Epilogue: attention blend + tanh + bias (C/D layout: row = r + 8*hh, col = N-tile*16 + lh)
    #pragma unroll
    for (int j = 0; j < 8; ++j) {
      const int col = (n_half * 8 + j) * 16 + lh;
      const float a  = attn[col];
      const float bb = biasc[col];
      #pragma unroll
      for (int r = 0; r < 8; ++r) {
        accL[j][r] = a * accL[j][r] + (1.f - a) * tanhf(accN[j][r]) + bb;
      }
    }

    // LayerNorm: per-row sum / sumsq over this wave's 128 cols, then cross-wave combine.
    float s[8], q[8];
    #pragma unroll
    for (int r = 0; r < 8; ++r) { s[r] = 0.f; q[r] = 0.f; }
    #pragma unroll
    for (int j = 0; j < 8; ++j)
      #pragma unroll
      for (int r = 0; r < 8; ++r) { const float x = accL[j][r]; s[r] += x; q[r] += x * x; }
    #pragma unroll
    for (int m = 1; m < 16; m <<= 1) {
      #pragma unroll
      for (int r = 0; r < 8; ++r) {
        s[r] += __shfl_xor(s[r], m, 32);
        q[r] += __shfl_xor(q[r], m, 32);
      }
    }
    if (lh == 0) {
      #pragma unroll
      for (int r = 0; r < 8; ++r) {
        const int row = m_tile * 16 + hh * 8 + r;
        redS[n_half * MT + row] = s[r];
        redQ[n_half * MT + row] = q[r];
      }
    }
    __syncthreads();

    float mu[8], rs[8];
    #pragma unroll
    for (int r = 0; r < 8; ++r) {
      const int row = m_tile * 16 + hh * 8 + r;
      const float tot = redS[row] + redS[MT + row];
      const float tq  = redQ[row] + redQ[MT + row];
      const float m_  = tot * (1.0f / FDIM);
      const float v_  = fmaxf(tq * (1.0f / FDIM) - m_ * m_, 0.f);
      mu[r] = m_;
      rs[r] = rsqrtf(v_ + LNEPS);
    }

    #pragma unroll
    for (int j = 0; j < 8; ++j) {
      const int col = (n_half * 8 + j) * 16 + lh;
      const float g  = gamma[col];
      const float be = beta[col];
      #pragma unroll
      for (int r = 0; r < 8; ++r) {
        const int row = rowbase + m_tile * 16 + hh * 8 + r;
        out[(size_t)row * FDIM + col] = (accL[j][r] - mu[r]) * rs[r] * g + be;
      }
    }
    __syncthreads();   // protect A/red LDS before next chunk
  }
}

// ---------------------------------------------------------------------------
extern "C" void kernel_launch(void* const* d_in, const int* in_sizes, int n_in,
                              void* d_out, int out_size, void* d_ws, size_t ws_size,
                              hipStream_t stream) {
  (void)in_sizes; (void)n_in; (void)out_size; (void)ws_size;
  const float* class_probs = (const float*)d_in[0];
  const float* w_lin  = (const float*)d_in[1];
  const float* w_non  = (const float*)d_in[2];
  const float* b_lin  = (const float*)d_in[3];
  const float* b_non  = (const float*)d_in[4];
  const float* f_attn = (const float*)d_in[5];
  const float* gamma  = (const float*)d_in[6];
  const float* beta   = (const float*)d_in[7];
  float* out = (float*)d_out;

  // Workspace layout: 4 split diag-weight matrices (bf16, transposed F x C) + attn + bias.
  uint16_t* wLH = (uint16_t*)d_ws;                 // 32768 elems
  uint16_t* wLL = wLH + FDIM * NCLS;
  uint16_t* wNH = wLL + FDIM * NCLS;
  uint16_t* wNL = wNH + FDIM * NCLS;
  float* attn  = (float*)(wNL + FDIM * NCLS);      // 256 floats
  float* biasc = attn + FDIM;                      // 256 floats

  prep_diag<<<NCLS, FDIM, 0, stream>>>(w_lin, w_non, wLH, wLL, wNH, wNL);
  prep_vec<<<1, FDIM, 0, stream>>>(f_attn, b_lin, b_non, attn, biasc);

  const int nchunks = BATCH / MT;                  // 1024
  const size_t smem = (size_t)(4 * FDIM * WST + 2 * MT * WST) * sizeof(uint16_t)
                    + (size_t)(4 * MT) * sizeof(float);   // 314,368 B <= 320 KB LDS
  dbnn_main<<<nchunks, 256, smem, stream>>>(class_probs, wLH, wLL, wNH, wNL,
                                            attn, biasc, gamma, beta, out, nchunks);
}
